// RGCN_3186865733924
// MI455X (gfx1250) — matrix operand order repaired and checked by reading, not compiled, not run on hardware
//
#include <hip/hip_runtime.h>

#define N_REVIEW  100000
#define N_PRODUCT 20000
#define N_NODES   (N_REVIEW + N_PRODUCT)
#define E_EDGES   1000000
#define HID  64
#define OUTD 32

typedef __attribute__((ext_vector_type(16))) _Float16 v16h;
typedef __attribute__((ext_vector_type(8)))  _Float16 v8h;
typedef __attribute__((ext_vector_type(2)))  _Float16 h2v;
typedef __attribute__((ext_vector_type(8)))  float    v8f;

// ---------------------------------------------------------------------------
// CDNA5 wave32 WMMA layouts (ISA 7.12.2):
//  A 16x32 f16 : lane l -> row M = l%16, half = l/16;
//                element e<8: K = half*8+e ; e>=8: K = 16+half*8+(e-8)
//  B 32x16 f16 : lane l -> col N = l%16 ; element e: K as for A (e<8: half*8+e ; e>=8: 16+half*8+(e-8))
//  C/D 16x16 f32 (v8f): lane l -> col N = l%16 ; element r: M = half*8+r
// ---------------------------------------------------------------------------

// A fragment from an f32 row (agg buffers / raw inputs), with per-row scale.
__device__ __forceinline__ v16h load_a_f32(const float* __restrict__ row,
                                           int kb, int half, float scale) {
  const float4* p = (const float4*)(row + kb + half * 8);
  float4 u0 = p[0];
  float4 u1 = p[1];
  const float4* q = (const float4*)(row + kb + 16 + half * 8);
  float4 u2 = q[0];
  float4 u3 = q[1];
  v16h a;
  a[0]  = (_Float16)(u0.x * scale); a[1]  = (_Float16)(u0.y * scale);
  a[2]  = (_Float16)(u0.z * scale); a[3]  = (_Float16)(u0.w * scale);
  a[4]  = (_Float16)(u1.x * scale); a[5]  = (_Float16)(u1.y * scale);
  a[6]  = (_Float16)(u1.z * scale); a[7]  = (_Float16)(u1.w * scale);
  a[8]  = (_Float16)(u2.x * scale); a[9]  = (_Float16)(u2.y * scale);
  a[10] = (_Float16)(u2.z * scale); a[11] = (_Float16)(u2.w * scale);
  a[12] = (_Float16)(u3.x * scale); a[13] = (_Float16)(u3.y * scale);
  a[14] = (_Float16)(u3.z * scale); a[15] = (_Float16)(u3.w * scale);
  return a;
}

// A fragment from an f16 row (h activations): two 16B loads, no conversion.
__device__ __forceinline__ v16h load_a_f16(const _Float16* __restrict__ row,
                                           int kb, int half) {
  v8h lo = *(const v8h*)(row + kb + half * 8);
  v8h hi = *(const v8h*)(row + kb + 16 + half * 8);
  v16h a;
#pragma unroll
  for (int i = 0; i < 8; ++i) { a[i] = lo[i]; a[i + 8] = hi[i]; }
  return a;
}

// ---------------------------------------------------------------------------
// Pack a row-major f32 weight W[K][NCOLS] into per-lane f16 B-fragments:
//   out[((c*(NCOLS/16)+t)*32 + l)*16 + e] = W[c*32 + (e>>3)*16 + (l>>4)*8 + (e&7)][t*16 + (l&15)]
// ---------------------------------------------------------------------------
__global__ void pack_b_kernel(const float* __restrict__ W, int K, int NCOLS,
                              _Float16* __restrict__ out) {
  int total = (K / 32) * (NCOLS / 16) * 32 * 16;
  int i = blockIdx.x * blockDim.x + threadIdx.x;
  if (i >= total) return;
  int e  = i & 15;
  int l  = (i >> 4) & 31;
  int ct = i >> 9;
  int nt = NCOLS / 16;
  int t  = ct % nt;
  int c  = ct / nt;
  int k  = c * 32 + (e >> 3) * 16 + (l >> 4) * 8 + (e & 7);
  int n  = t * 16 + (l & 15);
  out[i] = (_Float16)W[(size_t)k * NCOLS + n];
}

// ---------------------------------------------------------------------------
__global__ void zero_kernel(float4* __restrict__ p, long n4) {
  long i = (long)blockIdx.x * blockDim.x + threadIdx.x;
  if (i < n4) p[i] = make_float4(0.f, 0.f, 0.f, 0.f);
}

// ---------------------------------------------------------------------------
// Per-node-type linear: h[m][0:64] = x[m][0:K] @ W[K][64] + b   (h is f16)
// One wave computes a 16-row x 64-col strip (4 WMMA column tiles).
// ---------------------------------------------------------------------------
__global__ void node_linear_kernel(const float* __restrict__ x, int K,
                                   const _Float16* __restrict__ Bp,
                                   const float* __restrict__ b,
                                   _Float16* __restrict__ out, int nrows) {
  int wave = (int)((blockIdx.x * blockDim.x + threadIdx.x) >> 5);
  int lane = threadIdx.x & 31;
  int half = lane >> 4;
  int l16  = lane & 15;
  int mbase = wave * 16;
  if (mbase >= nrows) return;

  const float* arow = x + (size_t)(mbase + l16) * K;
  const v16h* Bf = (const v16h*)Bp;
  v8f acc[4] = {};
  for (int kb = 0; kb < K; kb += 32) {
    v16h a = load_a_f32(arow, kb, half, 1.0f);
    int cbase = (kb >> 5) * 4;
#pragma unroll
    for (int t = 0; t < 4; ++t) {
      v16h bb = Bf[(size_t)(cbase + t) * 32 + lane];
      acc[t] = __builtin_amdgcn_wmma_f32_16x16x32_f16(
          false, a, false, bb, (short)0, acc[t], false, false);
    }
  }
#pragma unroll
  for (int t = 0; t < 4; ++t) {
    int n = t * 16 + l16;
    float bias = b[n];
#pragma unroll
    for (int r = 0; r < 8; ++r) {
      int m = mbase + half * 8 + r;
      out[(size_t)m * HID + n] = (_Float16)(acc[t][r] + bias);
    }
  }
}

// ---------------------------------------------------------------------------
// Edge scatter: agg[dst] += widen(h[src]) (64 feats), cnt[dst] += 1 (optional)
// One wave per edge; each lane handles 2 f16 features (one b32 gather).
// ---------------------------------------------------------------------------
__global__ void scatter_kernel(const _Float16* __restrict__ h,
                               const int* __restrict__ src,
                               const int* __restrict__ dst,
                               float* __restrict__ agg,
                               float* __restrict__ cnt, int addCnt) {
  int tid  = blockIdx.x * blockDim.x + threadIdx.x;
  int e    = tid >> 5;
  if (e >= E_EDGES) return;
  int lane = tid & 31;
  int s = src[e];
  int d = dst[e];
  h2v v = ((const h2v*)(h + (size_t)s * HID))[lane];
  float* ap = agg + (size_t)d * HID + lane * 2;
  atomicAdd(ap,     (float)v[0]);
  atomicAdd(ap + 1, (float)v[1]);
  if (addCnt && lane == 0) atomicAdd(cnt + d, 1.0f);
}

// ---------------------------------------------------------------------------
// Fused RGCN layer: out = act( h@Wroot + mean0@Wrel0 + mean1@Wrel1 + bias )
// mean_r = agg_r / max(cnt_r,1) folded into A-fragment scale.
// One wave computes a full 16-row x NOUT strip; safe in-place (out == hin).
// ---------------------------------------------------------------------------
template <int NOUT, bool RELU, typename OutT>
__global__ void rgcn_layer_kernel(const _Float16* __restrict__ hin,
                                  const float* __restrict__ agg0,
                                  const float* __restrict__ agg1,
                                  const float* __restrict__ cnt0,
                                  const float* __restrict__ cnt1,
                                  const _Float16* __restrict__ Broot,
                                  const _Float16* __restrict__ Brel0,
                                  const _Float16* __restrict__ Brel1,
                                  const float* __restrict__ bias,
                                  OutT* __restrict__ out) {
  constexpr int NT = NOUT / 16;
  int wave = (int)((blockIdx.x * blockDim.x + threadIdx.x) >> 5);
  int lane = threadIdx.x & 31;
  int half = lane >> 4;
  int l16  = lane & 15;
  int mbase = wave * 16;
  if (mbase >= N_NODES) return;

  int ra = mbase + l16;
  float inv0 = 1.0f / fmaxf(cnt0[ra], 1.0f);
  float inv1 = 1.0f / fmaxf(cnt1[ra], 1.0f);

  v8f acc[NT] = {};

  // Root term: f16 A fragments straight from h, no conversion.
  const _Float16* hrow = hin + (size_t)ra * HID;
  const v16h* Bfr = (const v16h*)Broot;
#pragma unroll
  for (int c = 0; c < HID / 32; ++c) {
    v16h a = load_a_f16(hrow, c * 32, half);
#pragma unroll
    for (int t = 0; t < NT; ++t) {
      v16h bb = Bfr[(size_t)(c * NT + t) * 32 + lane];
      acc[t] = __builtin_amdgcn_wmma_f32_16x16x32_f16(
          false, a, false, bb, (short)0, acc[t], false, false);
    }
  }

  // Relation terms: f32 agg rows scaled by 1/max(cnt,1).
  const float* asrc[2] = { agg0 + (size_t)ra * HID, agg1 + (size_t)ra * HID };
  const v16h* Bs[2] = { (const v16h*)Brel0, (const v16h*)Brel1 };
  float scl[2] = { inv0, inv1 };
#pragma unroll
  for (int s = 0; s < 2; ++s) {
#pragma unroll
    for (int c = 0; c < HID / 32; ++c) {
      v16h a = load_a_f32(asrc[s], c * 32, half, scl[s]);
#pragma unroll
      for (int t = 0; t < NT; ++t) {
        v16h bb = Bs[s][(size_t)(c * NT + t) * 32 + lane];
        acc[t] = __builtin_amdgcn_wmma_f32_16x16x32_f16(
            false, a, false, bb, (short)0, acc[t], false, false);
      }
    }
  }

#pragma unroll
  for (int t = 0; t < NT; ++t) {
    int n = t * 16 + l16;
    float bv = bias[n];
#pragma unroll
    for (int r = 0; r < 8; ++r) {
      int m = mbase + half * 8 + r;
      float v = acc[t][r] + bv;
      if (RELU) v = fmaxf(v, 0.0f);
      out[(size_t)m * NOUT + n] = (OutT)v;
    }
  }
}

// ---------------------------------------------------------------------------
extern "C" void kernel_launch(void* const* d_in, const int* in_sizes, int n_in,
                              void* d_out, int out_size, void* d_ws, size_t ws_size,
                              hipStream_t stream) {
  const float* x_review      = (const float*)d_in[0];
  const float* x_product     = (const float*)d_in[1];
  const float* W_lin_review  = (const float*)d_in[2];
  const float* b_lin_review  = (const float*)d_in[3];
  const float* W_lin_product = (const float*)d_in[4];
  const float* b_lin_product = (const float*)d_in[5];
  const float* W_rel1        = (const float*)d_in[6];   // [2,64,64]
  const float* W_root1       = (const float*)d_in[7];
  const float* b1            = (const float*)d_in[8];
  const float* W_rel2        = (const float*)d_in[9];   // [2,64,32]
  const float* W_root2       = (const float*)d_in[10];
  const float* b2            = (const float*)d_in[11];
  const int*   src0          = (const int*)d_in[12];
  const int*   dst0          = (const int*)d_in[13];
  const int*   src1          = (const int*)d_in[14];
  const int*   dst1          = (const int*)d_in[15];
  float* out = (float*)d_out;

  // Workspace: h(f16) | agg0 | agg1 | cnt0 | cnt1 | packed f16 weights
  _Float16* h  = (_Float16*)d_ws;                       // N*HID f16 (15.36 MB)
  float* agg0  = (float*)(h + (size_t)N_NODES * HID);   // 16B-aligned
  float* agg1  = agg0 + (size_t)N_NODES * HID;
  float* cnt0  = agg1 + (size_t)N_NODES * HID;
  float* cnt1  = cnt0 + (size_t)N_NODES;
  _Float16* pk = (_Float16*)(cnt1 + (size_t)N_NODES);   // 32B-aligned

  _Float16* pWlinR  = pk;                 // 128x64 -> 8192 f16
  _Float16* pWlinP  = pWlinR  + 8192;     // 64x64  -> 4096
  _Float16* pWroot1 = pWlinP  + 4096;
  _Float16* pWrel10 = pWroot1 + 4096;
  _Float16* pWrel11 = pWrel10 + 4096;
  _Float16* pWroot2 = pWrel11 + 4096;     // 64x32  -> 2048
  _Float16* pWrel20 = pWroot2 + 2048;
  _Float16* pWrel21 = pWrel20 + 2048;

  const int TB = 256;                     // 8 waves / block
  const int wpb = TB / 32;

  // 0) pack weights into WMMA B-fragment layout (tiny)
  pack_b_kernel<<<(8192 + TB - 1) / TB, TB, 0, stream>>>(W_lin_review, 128, 64, pWlinR);
  pack_b_kernel<<<(4096 + TB - 1) / TB, TB, 0, stream>>>(W_lin_product, 64, 64, pWlinP);
  pack_b_kernel<<<(4096 + TB - 1) / TB, TB, 0, stream>>>(W_root1, 64, 64, pWroot1);
  pack_b_kernel<<<(4096 + TB - 1) / TB, TB, 0, stream>>>(W_rel1, 64, 64, pWrel10);
  pack_b_kernel<<<(4096 + TB - 1) / TB, TB, 0, stream>>>(W_rel1 + 64 * 64, 64, 64, pWrel11);
  pack_b_kernel<<<(2048 + TB - 1) / TB, TB, 0, stream>>>(W_root2, 64, 32, pWroot2);
  pack_b_kernel<<<(2048 + TB - 1) / TB, TB, 0, stream>>>(W_rel2, 64, 32, pWrel20);
  pack_b_kernel<<<(2048 + TB - 1) / TB, TB, 0, stream>>>(W_rel2 + 64 * 32, 64, 32, pWrel21);

  // 1) zero agg0|agg1|cnt0|cnt1 (contiguous)
  {
    long nz4 = ((long)N_NODES * HID * 2 + (long)N_NODES * 2) / 4;
    zero_kernel<<<(int)((nz4 + TB - 1) / TB), TB, 0, stream>>>((float4*)agg0, nz4);
  }

  // 2) per-node-type linears -> h (f16)
  node_linear_kernel<<<(N_REVIEW / 16 + wpb - 1) / wpb, TB, 0, stream>>>(
      x_review, 128, pWlinR, b_lin_review, h, N_REVIEW);
  node_linear_kernel<<<(N_PRODUCT / 16 + wpb - 1) / wpb, TB, 0, stream>>>(
      x_product, 64, pWlinP, b_lin_product, h + (size_t)N_REVIEW * HID, N_PRODUCT);

  // 3) edge aggregation (layer 1) + counts
  {
    int blocks = (E_EDGES * 32) / TB;
    scatter_kernel<<<blocks, TB, 0, stream>>>(h, src0, dst0, agg0, cnt0, 1);
    scatter_kernel<<<blocks, TB, 0, stream>>>(h, src1, dst1, agg1, cnt1, 1);
  }

  // 4) layer 1 (ReLU), in-place on h (f16 out)
  rgcn_layer_kernel<64, true, _Float16><<<(N_NODES / 16 + wpb - 1) / wpb, TB, 0, stream>>>(
      h, agg0, agg1, cnt0, cnt1, pWroot1, pWrel10, pWrel11, b1, h);

  // 5) re-zero agg0|agg1 (counts reused across layers)
  {
    long nz4 = ((long)N_NODES * HID * 2) / 4;
    zero_kernel<<<(int)((nz4 + TB - 1) / TB), TB, 0, stream>>>((float4*)agg0, nz4);
  }

  // 6) edge aggregation (layer 2)
  {
    int blocks = (E_EDGES * 32) / TB;
    scatter_kernel<<<blocks, TB, 0, stream>>>(h, src0, dst0, agg0, cnt0, 0);
    scatter_kernel<<<blocks, TB, 0, stream>>>(h, src1, dst1, agg1, cnt1, 0);
  }

  // 7) layer 2 -> d_out [N, 32] (f32)
  rgcn_layer_kernel<32, false, float><<<(N_NODES / 16 + wpb - 1) / wpb, TB, 0, stream>>>(
      h, agg0, agg1, cnt0, cnt1, pWroot2, pWrel20, pWrel21, b2, out);
}
